// NativeBitGPT_74620761801547
// MI455X (gfx1250) — matrix operand order repaired and checked by reading, not compiled, not run on hardware
//
#include <hip/hip_runtime.h>
#include <hip/hip_bf16.h>
#include <math.h>

typedef __attribute__((ext_vector_type(16))) __bf16 v16bf;
typedef __attribute__((ext_vector_type(8)))  float  v8f;
typedef int v4i __attribute__((__vector_size__(4 * sizeof(int))));

static constexpr int Vv = 50257, Ll = 8, Ee = 1024, Hh = 16, HDd = 64, FFf = 4096, Bb = 2, Tt = 1024;
static constexpr int MTOK = Bb * Tt;   // 2048 tokens

__device__ __forceinline__ __bf16 f2bf(float f) { return (__bf16)f; }

// ---------------------------------------------------------------------------
// Async global -> LDS copy (16B per lane), ASYNCcnt-tracked (ISA 08 §4)
// ---------------------------------------------------------------------------
__device__ __forceinline__ void async_ld16(const __bf16* g, __bf16* l)
{
#if __has_builtin(__builtin_amdgcn_global_load_async_to_lds_b128)
    __builtin_amdgcn_global_load_async_to_lds_b128(
        (__attribute__((address_space(1))) v4i*)(void*)g,
        (__attribute__((address_space(3))) v4i*)(void*)l, 0, 0);
#else
    unsigned l32 = (unsigned)(size_t)(__attribute__((address_space(3))) __bf16*)l;
    unsigned long long g64 = (unsigned long long)g;
    asm volatile("global_load_async_to_lds_b128 %0, %1, off"
                 :: "v"(l32), "v"(g64) : "memory");
#endif
}

#if __has_builtin(__builtin_amdgcn_s_wait_asynccnt)
#define WAIT_ASYNC(n) __builtin_amdgcn_s_wait_asynccnt(n)
#else
#define WAIT_ASYNC(n) asm volatile("s_wait_asynccnt %0" :: "i"(n) : "memory")
#endif

// ---------------------------------------------------------------------------
// GEMM: C[M,N] = A[M,K](bf16) * W[N,K](bf16)^T, fp32 accumulate.
// mode: 0=store, 1=C=Cin+acc (residual), 2=30*tanh(x/30) cap.
// Block: 256 threads = 8 waves (2 M x 4 N), block tile 64x256,
// wave tile 32x64 (2 A-frags x 4 B-frags -> 8 WMMAs / K-step of 32),
// double-buffered async global->LDS staging (5 b128 per wave per tile).
// ---------------------------------------------------------------------------
__global__ __launch_bounds__(256)
void gemm_awt(const __bf16* __restrict__ A, const __bf16* __restrict__ W,
              const float* __restrict__ Cin, float* __restrict__ C,
              int M, int N, int K, int mode)
{
    __shared__ __align__(16) __bf16 As[2][64][32];
    __shared__ __align__(16) __bf16 Ws[2][256][32];

    const int tid  = threadIdx.x;
    const int lane = tid & 31;
    const int wid  = tid >> 5;
    const int wm   = wid & 1;           // 0..1  (M)
    const int wn   = wid >> 1;          // 0..3  (N)
    const int mbase = blockIdx.y * 64;
    const int nbase = blockIdx.x * 256;

    v8f acc[8];
    #pragma unroll
    for (int i = 0; i < 8; ++i)
        #pragma unroll
        for (int r = 0; r < 8; ++r) acc[i][r] = 0.f;

    // stage one 64x32 A tile + 256x32 W tile: 5 async b128 per wave
    auto issue = [&](int buf, int k0) {
        int ar = tid >> 2, aseg = (tid & 3) * 8;
        async_ld16(A + (size_t)(mbase + ar) * K + k0 + aseg, &As[buf][ar][aseg]);
        #pragma unroll
        for (int it = 0; it < 4; ++it) {
            int c  = tid + it * 256;
            int wr = c >> 2, wseg = (c & 3) * 8;
            int n  = nbase + wr;
            if (n >= N) n = N - 1;      // clamp: garbage cols masked at store
            async_ld16(W + (size_t)n * K + k0 + wseg, &Ws[buf][wr][wseg]);
        }
    };

    const int nk = K >> 5;
    issue(0, 0);
    for (int kt = 0; kt < nk; ++kt) {
        const int buf = kt & 1;
        if (kt + 1 < nk) { issue(buf ^ 1, (kt + 1) * 32); WAIT_ASYNC(5); }
        else             { WAIT_ASYNC(0); }
        __syncthreads();

        // A fragments: lane row = lane&15; K halves per ISA 7.12.2
        v16bf afr[2];
        #pragma unroll
        for (int mi = 0; mi < 2; ++mi) {
            int row = wm * 32 + mi * 16 + (lane & 15);
            int kb  = (lane >> 4) * 8;
            #pragma unroll
            for (int j = 0; j < 8; ++j) {
                afr[mi][j]     = As[buf][row][kb + j];
                afr[mi][8 + j] = As[buf][row][16 + kb + j];
            }
        }
        // B fragments: lane col = lane&15; K halves per ISA 7.12.2 (same as A)
        v16bf bfr[4];
        #pragma unroll
        for (int ni = 0; ni < 4; ++ni) {
            int n  = wn * 64 + ni * 16 + (lane & 15);
            int kb = (lane >> 4) * 8;
            #pragma unroll
            for (int j = 0; j < 8; ++j) {
                bfr[ni][j]     = Ws[buf][n][kb + j];
                bfr[ni][8 + j] = Ws[buf][n][16 + kb + j];
            }
        }
        #pragma unroll
        for (int mi = 0; mi < 2; ++mi)
            #pragma unroll
            for (int ni = 0; ni < 4; ++ni)
                acc[mi * 4 + ni] = __builtin_amdgcn_wmma_f32_16x16x32_bf16(
                    false, afr[mi], false, bfr[ni], (short)0,
                    acc[mi * 4 + ni], false, false);
        __syncthreads();   // reads done before next issue overwrites this buf
    }

    // epilogue: C layout => lane holds column, 8 rows in VGPR slots
    #pragma unroll
    for (int mi = 0; mi < 2; ++mi)
        #pragma unroll
        for (int ni = 0; ni < 4; ++ni) {
            int col = nbase + wn * 64 + ni * 16 + (lane & 15);
            if (col >= N) continue;
            int row0 = mbase + wm * 32 + mi * 16 + ((lane >> 4) * 8);
            #pragma unroll
            for (int r = 0; r < 8; ++r) {
                size_t idx = (size_t)(row0 + r) * N + col;
                float v = acc[mi * 4 + ni][r];
                if (mode == 1)      v += Cin[idx];
                else if (mode == 2) v = 30.f * tanhf(v * (1.f / 30.f));
                C[idx] = v;
            }
        }
}

// ---------------------------------------------------------------------------
// Flash attention: one wave per 16-row Q tile of one (b,h).
// ---------------------------------------------------------------------------
__global__ __launch_bounds__(32)
void attn_k(const float* __restrict__ Q, const float* __restrict__ Km,
            const float* __restrict__ Vm, __bf16* __restrict__ O)
{
    __shared__ __bf16 Pl[16][32];

    const int lane  = threadIdx.x;
    const int qt    = blockIdx.x & (Tt / 16 - 1);
    const int bh    = blockIdx.x / (Tt / 16);
    const int b     = bh / Hh;
    const int h     = bh % Hh;
    const int qbase = qt * 16;
    const size_t hb = (size_t)bh * Tt * HDd;
    const float scale = 0.125f;
    const int rowoff = (lane >> 4) * 8;

    v16bf qa[2];
    {
        int row = qbase + (lane & 15);
        int kb  = (lane >> 4) * 8;
        const float* qr = Q + hb + (size_t)row * HDd;
        #pragma unroll
        for (int half = 0; half < 2; ++half)
            #pragma unroll
            for (int j = 0; j < 8; ++j) {
                qa[half][j]     = f2bf(qr[half * 32 + kb + j]);
                qa[half][8 + j] = f2bf(qr[half * 32 + 16 + kb + j]);
            }
    }

    v8f oacc[4];
    float mrow[8], lrow[8];
    #pragma unroll
    for (int i = 0; i < 4; ++i)
        #pragma unroll
        for (int r = 0; r < 8; ++r) oacc[i][r] = 0.f;
    #pragma unroll
    for (int r = 0; r < 8; ++r) { mrow[r] = -1e30f; lrow[r] = 0.f; }

    const int nkt = (qbase + 15) / 32 + 1;
    for (int kt = 0; kt < nkt; ++kt) {
        const int kbase = kt * 32;
        v8f s[2];
        #pragma unroll
        for (int nt = 0; nt < 2; ++nt) {
            v16bf kf0, kf1;
            int key = kbase + nt * 16 + (lane & 15);
            int kb  = (lane >> 4) * 8;
            const float* kr = Km + hb + (size_t)key * HDd;
            #pragma unroll
            for (int j = 0; j < 8; ++j) {
                kf0[j]     = f2bf(kr[kb + j]);
                kf0[8 + j] = f2bf(kr[16 + kb + j]);
                kf1[j]     = f2bf(kr[32 + kb + j]);
                kf1[8 + j] = f2bf(kr[48 + kb + j]);
            }
            v8f z;
            #pragma unroll
            for (int r = 0; r < 8; ++r) z[r] = 0.f;
            z     = __builtin_amdgcn_wmma_f32_16x16x32_bf16(false, qa[0], false, kf0, (short)0, z, false, false);
            s[nt] = __builtin_amdgcn_wmma_f32_16x16x32_bf16(false, qa[1], false, kf1, (short)0, z, false, false);
        }

        float p0[8], p1[8];
        #pragma unroll
        for (int r = 0; r < 8; ++r) {
            int trow = qbase + rowoff + r;
            int key0 = kbase + (lane & 15);
            float s0 = s[0][r] * scale; if (key0 > trow)      s0 = -1e30f;
            float s1 = s[1][r] * scale; if (key0 + 16 > trow) s1 = -1e30f;
            float mx = fmaxf(s0, s1);
            #pragma unroll
            for (int msk = 1; msk < 16; msk <<= 1) mx = fmaxf(mx, __shfl_xor(mx, msk, 32));
            float mnew  = fmaxf(mrow[r], mx);
            float alpha = expf(mrow[r] - mnew);
            float e0 = expf(s0 - mnew), e1 = expf(s1 - mnew);
            float ps = e0 + e1;
            #pragma unroll
            for (int msk = 1; msk < 16; msk <<= 1) ps += __shfl_xor(ps, msk, 32);
            lrow[r] = lrow[r] * alpha + ps;
            mrow[r] = mnew;
            p0[r] = e0; p1[r] = e1;
            #pragma unroll
            for (int nt2 = 0; nt2 < 4; ++nt2) oacc[nt2][r] *= alpha;
        }

        #pragma unroll
        for (int r = 0; r < 8; ++r) {
            Pl[rowoff + r][lane & 15]        = f2bf(p0[r]);
            Pl[rowoff + r][16 + (lane & 15)] = f2bf(p1[r]);
        }
        __syncthreads();
        v16bf pa;
        {
            int row = lane & 15;
            int kb  = (lane >> 4) * 16;
            #pragma unroll
            for (int j = 0; j < 16; ++j) pa[j] = Pl[row][kb + j];
        }
        #pragma unroll
        for (int nt2 = 0; nt2 < 4; ++nt2) {
            v16bf vf;
            int hd  = nt2 * 16 + (lane & 15);
            int kk0 = kbase + (lane >> 4) * 16;
            #pragma unroll
            for (int j = 0; j < 16; ++j)
                vf[j] = f2bf(Vm[hb + (size_t)(kk0 + j) * HDd + hd]);
            oacc[nt2] = __builtin_amdgcn_wmma_f32_16x16x32_bf16(
                false, pa, false, vf, (short)0, oacc[nt2], false, false);
        }
        __syncthreads();
    }

    #pragma unroll
    for (int nt2 = 0; nt2 < 4; ++nt2) {
        int e = h * HDd + nt2 * 16 + (lane & 15);
        #pragma unroll
        for (int r = 0; r < 8; ++r) {
            int t = qbase + rowoff + r;
            O[(size_t)(b * Tt + t) * Ee + e] = f2bf(oacc[nt2][r] / lrow[r]);
        }
    }
}

// ---------------------------------------------------------------------------
// Elementwise / small kernels
// ---------------------------------------------------------------------------
__global__ void cvt_bf16_k(const float* __restrict__ s, __bf16* __restrict__ d, size_t n)
{
    size_t i = ((size_t)blockIdx.x * blockDim.x + threadIdx.x) * 4;
    if (i + 3 < n) {
        float4 v = *(const float4*)(s + i);
        d[i]     = f2bf(v.x); d[i + 1] = f2bf(v.y);
        d[i + 2] = f2bf(v.z); d[i + 3] = f2bf(v.w);
    } else {
        for (; i < n; ++i) d[i] = f2bf(s[i]);
    }
}

__global__ void embed_k(const int* __restrict__ idx, const float* __restrict__ emb,
                        float* __restrict__ x, float* __restrict__ x0)
{
    int row = blockIdx.x;
    int tok = idx[row];
    const float* src = emb + (size_t)tok * Ee;
    for (int e = threadIdx.x; e < Ee; e += blockDim.x) {
        float v = src[e];
        x [(size_t)row * Ee + e] = v;
        x0[(size_t)row * Ee + e] = v;
    }
}

__global__ __launch_bounds__(256)
void rmsnorm_k(const float* __restrict__ x, const float* __restrict__ w,
               __bf16* __restrict__ out)
{
    __shared__ float red[256];
    int row = blockIdx.x;
    const float* xr = x + (size_t)row * Ee;
    float s = 0.f;
    for (int e = threadIdx.x; e < Ee; e += 256) { float v = xr[e]; s += v * v; }
    red[threadIdx.x] = s;
    __syncthreads();
    for (int off = 128; off > 0; off >>= 1) {
        if (threadIdx.x < off) red[threadIdx.x] += red[threadIdx.x + off];
        __syncthreads();
    }
    float norm = rsqrtf(red[0] * (1.f / Ee) + 1e-6f);
    for (int e = threadIdx.x; e < Ee; e += 256)
        out[(size_t)row * Ee + e] = f2bf(xr[e] * norm * w[e]);
}

__global__ __launch_bounds__(64)
void qkvpost_k(const float* __restrict__ qkv, const __bf16* __restrict__ hn,
               const float* __restrict__ x0, const float* __restrict__ veg,
               float* __restrict__ Q, float* __restrict__ K2, float* __restrict__ V2,
               int use_gate)
{
    int d   = threadIdx.x;
    int bid = blockIdx.x;
    int h   = bid % Hh;
    int t   = (bid / Hh) % Tt;
    int b   = bid / (Hh * Tt);
    size_t tb = (size_t)(b * Tt + t);
    const float* base = qkv + tb * (3 * Ee);
    float qv = base[h * HDd + d];
    float kv = base[Ee + h * HDd + d];
    float vv = base[2 * Ee + h * HDd + d];

    int   j   = d & 31;
    float inv = expf(-(float)(2 * j) * (1.f / 64.f) * 9.210340371976184f); // ln(1e4)
    float ang = (float)t * inv;
    float c = cosf(ang), sn = sinf(ang);
    float qp = __shfl_xor(qv, 1, 32);
    float kp = __shfl_xor(kv, 1, 32);
    float qrot = (d & 1) ? qp : -qp;
    float krot = (d & 1) ? kp : -kp;
    qv = qv * c + qrot * sn;
    kv = kv * c + krot * sn;

    if (use_gate) {
        const __bf16* hr = hn + tb * Ee;
        const float*  wg = veg + h * 32;
        float dot = 0.f;
        #pragma unroll
        for (int cc = 0; cc < 32; ++cc) dot += (float)hr[cc] * wg[cc];
        float g = 2.f / (1.f + expf(-dot));
        vv += g * x0[tb * Ee + h * HDd + d];
    }
    size_t o = ((size_t)(b * Hh + h) * Tt + t) * HDd + d;
    Q[o] = qv; K2[o] = kv; V2[o] = vv;
}

__global__ void silumul_k(const float* __restrict__ g, const float* __restrict__ u,
                          __bf16* __restrict__ out, size_t n)
{
    size_t i = (size_t)blockIdx.x * blockDim.x + threadIdx.x;
    if (i < n) {
        float gv = g[i];
        out[i] = f2bf(gv / (1.f + expf(-gv)) * u[i]);
    }
}

// ---------------------------------------------------------------------------
extern "C" void kernel_launch(void* const* d_in, const int* in_sizes, int n_in,
                              void* d_out, int out_size, void* d_ws, size_t ws_size,
                              hipStream_t stream)
{
    (void)in_sizes; (void)n_in; (void)out_size; (void)ws_size;
    const int*   idx     = (const int*)d_in[0];
    const float* tok_emb = (const float*)d_in[1];
    const float* ln1_w   = (const float*)d_in[2];
    const float* qkv_w   = (const float*)d_in[3];
    const float* out_w   = (const float*)d_in[4];
    const float* veg_w   = (const float*)d_in[5];
    const float* ln2_w   = (const float*)d_in[6];
    const float* wg_w    = (const float*)d_in[7];
    const float* wu_w    = (const float*)d_in[8];
    const float* wd_w    = (const float*)d_in[9];
    const float* lnf_w   = (const float*)d_in[10];
    float* out = (float*)d_out;

    // ---- workspace layout: fp32 region, then bf16 region -----------------
    float* ws = (float*)d_ws;
    size_t off = 0;
    auto allocf = [&](size_t n) { float* p = ws + off; off += (n + 7) & ~(size_t)7; return p; };
    float* x    = allocf((size_t)MTOK * Ee);
    float* x0   = allocf((size_t)MTOK * Ee);
    float* qkv  = allocf((size_t)MTOK * 3 * Ee);
    float* qb   = allocf((size_t)MTOK * Ee);
    float* kb   = allocf((size_t)MTOK * Ee);
    float* vb   = allocf((size_t)MTOK * Ee);
    float* gb   = allocf((size_t)MTOK * FFf);
    float* ub   = allocf((size_t)MTOK * FFf);

    __bf16* bbase = (__bf16*)(ws + off);
    size_t boff = 0;
    auto allocb = [&](size_t n) { __bf16* p = bbase + boff; boff += (n + 7) & ~(size_t)7; return p; };
    __bf16* hbuf   = allocb((size_t)MTOK * Ee);
    __bf16* ob     = allocb((size_t)MTOK * Ee);
    __bf16* gub    = allocb((size_t)MTOK * FFf);
    __bf16* qkvw_b = allocb((size_t)Ll * 3 * Ee * Ee);
    __bf16* outw_b = allocb((size_t)Ll * Ee * Ee);
    __bf16* wg_b   = allocb((size_t)Ll * FFf * Ee);
    __bf16* wu_b   = allocb((size_t)Ll * FFf * Ee);
    __bf16* wd_b   = allocb((size_t)Ll * Ee * FFf);
    __bf16* emb_b  = allocb((size_t)Vv * Ee);

    // ---- one-shot weight conversion fp32 -> bf16 (L2-resident weights) ---
    auto cvt = [&](const float* s, __bf16* d, size_t n) {
        cvt_bf16_k<<<(unsigned)((n / 4 + 255) / 256), 256, 0, stream>>>(s, d, n);
    };
    cvt(qkv_w,   qkvw_b, (size_t)Ll * 3 * Ee * Ee);
    cvt(out_w,   outw_b, (size_t)Ll * Ee * Ee);
    cvt(wg_w,    wg_b,   (size_t)Ll * FFf * Ee);
    cvt(wu_w,    wu_b,   (size_t)Ll * FFf * Ee);
    cvt(wd_w,    wd_b,   (size_t)Ll * Ee * FFf);
    cvt(tok_emb, emb_b,  (size_t)Vv * Ee);

    auto ggrid = [](int M, int N) { return dim3((unsigned)((N + 255) / 256), (unsigned)((M + 63) / 64)); };

    embed_k<<<MTOK, 256, 0, stream>>>(idx, tok_emb, x, x0);

    for (int i = 0; i < Ll; ++i) {
        rmsnorm_k<<<MTOK, 256, 0, stream>>>(x, ln1_w + (size_t)i * Ee, hbuf);
        gemm_awt<<<ggrid(MTOK, 3 * Ee), 256, 0, stream>>>(
            hbuf, qkvw_b + (size_t)i * 3 * Ee * Ee, nullptr, qkv, MTOK, 3 * Ee, Ee, 0);
        qkvpost_k<<<Bb * Tt * Hh, 64, 0, stream>>>(
            qkv, hbuf, x0, veg_w + (size_t)i * Hh * 32, qb, kb, vb, (i & 1));
        attn_k<<<Bb * Hh * (Tt / 16), 32, 0, stream>>>(qb, kb, vb, ob);
        gemm_awt<<<ggrid(MTOK, Ee), 256, 0, stream>>>(
            ob, outw_b + (size_t)i * Ee * Ee, x, x, MTOK, Ee, Ee, 1);
        rmsnorm_k<<<MTOK, 256, 0, stream>>>(x, ln2_w + (size_t)i * Ee, hbuf);
        gemm_awt<<<ggrid(MTOK, FFf), 256, 0, stream>>>(
            hbuf, wg_b + (size_t)i * FFf * Ee, nullptr, gb, MTOK, FFf, Ee, 0);
        gemm_awt<<<ggrid(MTOK, FFf), 256, 0, stream>>>(
            hbuf, wu_b + (size_t)i * FFf * Ee, nullptr, ub, MTOK, FFf, Ee, 0);
        size_t nel = (size_t)MTOK * FFf;
        silumul_k<<<(unsigned)((nel + 255) / 256), 256, 0, stream>>>(gb, ub, gub, nel);
        gemm_awt<<<ggrid(MTOK, Ee), 256, 0, stream>>>(
            gub, wd_b + (size_t)i * Ee * FFf, x, x, MTOK, Ee, FFf, 1);
    }

    rmsnorm_k<<<MTOK, 256, 0, stream>>>(x, lnf_w, hbuf);
    gemm_awt<<<ggrid(MTOK, Vv), 256, 0, stream>>>(
        hbuf, emb_b, nullptr, out, MTOK, Vv, Ee, 2);
}
